// Transformer_40106404610090
// MI455X (gfx1250) — compile-verified, matrix operand order checked by reading, not run on hardware
//
#include <hip/hip_runtime.h>
#include <hip/hip_bf16.h>
#include <cstdint>

#define DM   512
#define NH   8
#define HDM  64
#define DFFN 2048
#define NBATCH 2
#define TT   1024
#define NLAY 6

typedef __bf16 bf16;
typedef __attribute__((ext_vector_type(8)))  __bf16 v8bf;
typedef __attribute__((ext_vector_type(16))) __bf16 v16bf;
typedef __attribute__((ext_vector_type(8)))  float  v8f;

enum { FLAG_F32 = 1, FLAG_BF = 2, FLAG_RELU = 4 };
enum { BM_NONE = 0, BM_QK = 1, BM_PV = 2 };
enum { OM_LIN = 0, OM_VT = 1, OM_CTX = 2 };

// ---------------------------------------------------------------------------
// bf16 WMMA fragment load. ISA 16-bit A/B layout (16x32 / 32x16):
// lanes 0-15 hold K=0..7 and 16..23, lanes 16-31 hold K=8..15 and 24..31,
// element pairs contiguous -> two 16B vector loads per fragment.
// ---------------------------------------------------------------------------
__device__ __forceinline__ v16bf load_frag(const bf16* __restrict__ p, int lane) {
  const bf16* q = p + ((lane >> 4) << 3);
  v8bf lo = *(const v8bf*)q;
  v8bf hi = *(const v8bf*)(q + 16);
  return __builtin_shufflevector(lo, hi, 0,1,2,3,4,5,6,7,8,9,10,11,12,13,14,15);
}

// ---------------------------------------------------------------------------
// General bf16 GEMM: C[m,n] = sum_k A[m,k] * Bop[k,n] (+bias, relu)
// A: bf16 row-major [M,K] (lda).  Bp: bf16 packed as [N,K] (ldb = row stride).
// Each wave computes a 16x64 tile (A fragment reused over 4 WMMAs).
// Block = 128 threads = 4 waves stacked along M -> 64x64 block tile.
// ---------------------------------------------------------------------------
__global__ __launch_bounds__(128)
void gemm_wmma(const bf16* __restrict__ A, int lda,
               const bf16* __restrict__ Bp, int ldb,
               const float* __restrict__ bias,
               float* __restrict__ Cf, bf16* __restrict__ Cb, int ldc,
               int K, int flags, int bmode, int omode)
{
  const int z = blockIdx.z;
  long aoff = 0, boff = 0, coff = 0;
  if (bmode == BM_QK) {                 // per-(b,h) q/k slices of [B,T,D] tensors
    int b = z >> 3, h = z & 7;
    aoff = boff = (long)b * TT * DM + h * HDM;
    coff = (long)z * TT * TT;
  } else if (bmode == BM_PV) {          // probs [B,H,T,S] x vT [B,H,HD,S]
    aoff = (long)z * TT * TT;
    boff = (long)z * HDM * TT;
  }
  const int lane = threadIdx.x & 31;
  const int wave = threadIdx.x >> 5;
  const int m0 = blockIdx.y * 64 + wave * 16;
  const int n0 = blockIdx.x * 64;
  const int lrow = lane & 15;

  const bf16* arow = A + aoff + (long)(m0 + lrow) * lda;
  const bf16* br0  = Bp + boff + (long)(n0 + lrow) * ldb;
  const bf16* br1  = br0 + (long)16 * ldb;
  const bf16* br2  = br0 + (long)32 * ldb;
  const bf16* br3  = br0 + (long)48 * ldb;

  v8f acc[4] = {};
  for (int k = 0; k < K; k += 32) {
    v16bf af  = load_frag(arow + k, lane);
    v16bf bf0 = load_frag(br0 + k, lane);
    v16bf bf1 = load_frag(br1 + k, lane);
    v16bf bf2 = load_frag(br2 + k, lane);
    v16bf bf3 = load_frag(br3 + k, lane);
    acc[0] = __builtin_amdgcn_wmma_f32_16x16x32_bf16(false, af, false, bf0, (short)0, acc[0], false, false);
    acc[1] = __builtin_amdgcn_wmma_f32_16x16x32_bf16(false, af, false, bf1, (short)0, acc[1], false, false);
    acc[2] = __builtin_amdgcn_wmma_f32_16x16x32_bf16(false, af, false, bf2, (short)0, acc[2], false, false);
    acc[3] = __builtin_amdgcn_wmma_f32_16x16x32_bf16(false, af, false, bf3, (short)0, acc[3], false, false);
  }

  // C/D layout: VGPR r -> M = r (lanes 0-15) or r+8 (lanes 16-31); N = lane&15
  const int mb = m0 + ((lane >> 4) << 3);
#pragma unroll
  for (int t = 0; t < 4; ++t) {
    const int n = n0 + t * 16 + lrow;
    const float bv = bias ? bias[n] : 0.f;
#pragma unroll
    for (int r = 0; r < 8; ++r) {
      const int m = mb + r;
      float v = acc[t][r] + bv;
      if (flags & FLAG_RELU) v = fmaxf(v, 0.f);
      if (omode == OM_LIN) {
        const long idx = coff + (long)m * ldc + n;
        if (flags & FLAG_F32) Cf[idx] = v;
        if (flags & FLAG_BF)  Cb[idx] = (bf16)v;
      } else if (omode == OM_VT) {       // v-proj -> vT[b,h,e,s]
        const int b = m >> 10, s = m & (TT - 1);
        const int h = n >> 6,  e = n & (HDM - 1);
        Cb[((long)((b * NH + h) * HDM + e)) * TT + s] = (bf16)v;
      } else {                            // OM_CTX: context -> [B,T,H*HD]
        const int b = z >> 3, h = z & 7;
        Cb[(long)(b * TT + m) * DM + h * HDM + n] = (bf16)v;
      }
    }
  }
}

// ---------------------------------------------------------------------------
// Weight repack kernels (fp32 -> bf16, into [N,K] "B-transposed" layout)
// ---------------------------------------------------------------------------
__global__ __launch_bounds__(256)
void pack_headstacked(const float* __restrict__ w, bf16* __restrict__ o)
{ // w: [H, D, HD]  -> o[n*D + k], n = h*HD+e, k = d
  int i = blockIdx.x * 256 + threadIdx.x;     // over D*D
  int n = i >> 9, k = i & (DM - 1);
  int h = n >> 6, e = n & (HDM - 1);
  o[i] = (bf16)w[((long)h * DM + k) * HDM + e];
}

__global__ __launch_bounds__(256)
void pack_transpose(const float* __restrict__ w, bf16* __restrict__ o, int K, int N)
{ // w: [K,N] row-major -> o[n*K + k]
  int i = blockIdx.x * 256 + threadIdx.x;     // over N*K
  int n = i / K, k = i - n * K;
  o[i] = (bf16)w[(long)k * N + n];
}

__global__ __launch_bounds__(256)
void cvt_copy(const float* __restrict__ s, float* __restrict__ df,
              bf16* __restrict__ db, int n)
{
  int i = blockIdx.x * 256 + threadIdx.x;
  if (i < n) { float v = s[i]; df[i] = v; db[i] = (bf16)v; }
}

// ---------------------------------------------------------------------------
// Softmax over S=1024 with 1/sqrt(HD) scale and key-padding mask
// row = (b*H + h)*T + t ; writes bf16 probabilities
// ---------------------------------------------------------------------------
__global__ __launch_bounds__(256)
void softmax_bf16(const float* __restrict__ S, bf16* __restrict__ P,
                  const int* __restrict__ mask)
{
  const int row = blockIdx.x;
  const int b = row >> 13;                    // row / (H*T)
  const long base = (long)row * TT;
  const int tid = threadIdx.x;
  __shared__ float red[256];

  float v[4];
#pragma unroll
  for (int j = 0; j < 4; ++j) {
    int s = tid + j * 256;
    float x = S[base + s] * 0.125f;
    v[j] = (mask[b * TT + s] == 0) ? -INFINITY : x;
  }
  float mx = fmaxf(fmaxf(v[0], v[1]), fmaxf(v[2], v[3]));
  red[tid] = mx; __syncthreads();
  for (int s2 = 128; s2 > 0; s2 >>= 1) {
    if (tid < s2) red[tid] = fmaxf(red[tid], red[tid + s2]);
    __syncthreads();
  }
  mx = red[0]; __syncthreads();

  float sum = 0.f;
#pragma unroll
  for (int j = 0; j < 4; ++j) { v[j] = __expf(v[j] - mx); sum += v[j]; }
  red[tid] = sum; __syncthreads();
  for (int s2 = 128; s2 > 0; s2 >>= 1) {
    if (tid < s2) red[tid] += red[tid + s2];
    __syncthreads();
  }
  const float inv = 1.f / red[0];
#pragma unroll
  for (int j = 0; j < 4; ++j)
    P[base + tid + j * 256] = (bf16)(v[j] * inv);
}

// ---------------------------------------------------------------------------
// Fused residual (add or mul) + LayerNorm over D=512; writes f32 + bf16
// ---------------------------------------------------------------------------
__global__ __launch_bounds__(256)
void ln_fused(const float* __restrict__ X, const float* __restrict__ U,
              const float* __restrict__ g, const float* __restrict__ beta,
              float* __restrict__ Of, bf16* __restrict__ Ob, int mulMode)
{
  const long base = (long)blockIdx.x * DM;
  const int tid = threadIdx.x;
  __shared__ float red[256];

  float a0 = X[base + tid],       a1 = X[base + tid + 256];
  float u0 = U[base + tid],       u1 = U[base + tid + 256];
  float v0 = mulMode ? a0 * u0 : a0 + u0;
  float v1 = mulMode ? a1 * u1 : a1 + u1;

  red[tid] = v0 + v1; __syncthreads();
  for (int s = 128; s > 0; s >>= 1) { if (tid < s) red[tid] += red[tid + s]; __syncthreads(); }
  const float mu = red[0] * (1.f / DM); __syncthreads();

  float d0 = v0 - mu, d1 = v1 - mu;
  red[tid] = d0 * d0 + d1 * d1; __syncthreads();
  for (int s = 128; s > 0; s >>= 1) { if (tid < s) red[tid] += red[tid + s]; __syncthreads(); }
  const float inv = rsqrtf(red[0] * (1.f / DM) + 1e-5f); __syncthreads();

  float y0 = g[tid]       * d0 * inv + beta[tid];
  float y1 = g[tid + 256] * d1 * inv + beta[tid + 256];
  Of[base + tid] = y0;        Of[base + tid + 256] = y1;
  Ob[base + tid] = (bf16)y0;  Ob[base + tid + 256] = (bf16)y1;
}

// ---------------------------------------------------------------------------
// Host side
// ---------------------------------------------------------------------------
namespace {
struct MHAp { const float *wq,*bq,*wk,*bk,*wv,*bv,*wo,*bo; };
struct FFNp { const float *w1,*b1,*w2,*b2; };
struct LNp  { const float *g,*b; };
struct EncL { MHAp mha; FFNp ffn; LNp ln1, ln2; };
struct DecL { MHAp mha1, mha2; FFNp ffn; LNp ln1, ln2, ln3; };
}

extern "C" void kernel_launch(void* const* d_in, const int* in_sizes, int n_in,
                              void* d_out, int out_size, void* d_ws, size_t ws_size,
                              hipStream_t stream)
{
  const int ACT_N = NBATCH * TT * DM;                 // 1,048,576
  auto F = [&](int i) { return (const float*)d_in[i]; };

  // --- detect flattening order of the params pytree from in_sizes ---
  bool topSorted, sortedInner;
  int p0;
  if (n_in >= 5 && in_sizes[0] == ACT_N) { topSorted = false; p0 = 4; }
  else                                   { topSorted = true;  p0 = 0; }
  if (topSorted) sortedInner = true;                       // jax tree_leaves: fully sorted
  else sortedInner = (in_sizes[p0] != NH * DM * HDM);      // insertion order starts with wq
  const bool decFirst = sortedInner;                       // sorted dict: "dec" < "enc"

  int idx = p0;
  auto getMHA = [&]() {
    MHAp m;
    if (sortedInner) { m.bk=F(idx++); m.bo=F(idx++); m.bq=F(idx++); m.bv=F(idx++);
                       m.wk=F(idx++); m.wo=F(idx++); m.wq=F(idx++); m.wv=F(idx++); }
    else             { m.wq=F(idx++); m.bq=F(idx++); m.wk=F(idx++); m.bk=F(idx++);
                       m.wv=F(idx++); m.bv=F(idx++); m.wo=F(idx++); m.bo=F(idx++); }
    return m;
  };
  auto getFFN = [&]() {
    FFNp f;
    if (sortedInner) { f.b1=F(idx++); f.b2=F(idx++); f.w1=F(idx++); f.w2=F(idx++); }
    else             { f.w1=F(idx++); f.b1=F(idx++); f.w2=F(idx++); f.b2=F(idx++); }
    return f;
  };
  auto getLN = [&]() {
    LNp l;
    if (sortedInner) { l.b=F(idx++); l.g=F(idx++); }
    else             { l.g=F(idx++); l.b=F(idx++); }
    return l;
  };
  auto getEnc = [&]() {
    EncL e;
    if (sortedInner) { e.ffn=getFFN(); e.ln1=getLN(); e.ln2=getLN(); e.mha=getMHA(); }
    else             { e.mha=getMHA(); e.ffn=getFFN(); e.ln1=getLN(); e.ln2=getLN(); }
    return e;
  };
  auto getDec = [&]() {
    DecL d;
    if (sortedInner) { d.ffn=getFFN(); d.ln1=getLN(); d.ln2=getLN(); d.ln3=getLN();
                       d.mha1=getMHA(); d.mha2=getMHA(); }
    else             { d.mha1=getMHA(); d.mha2=getMHA(); d.ffn=getFFN();
                       d.ln1=getLN(); d.ln2=getLN(); d.ln3=getLN(); }
    return d;
  };

  EncL enc[NLAY]; DecL dec[NLAY];
  if (decFirst) { for (int l=0;l<NLAY;++l) dec[l]=getDec(); for (int l=0;l<NLAY;++l) enc[l]=getEnc(); }
  else          { for (int l=0;l<NLAY;++l) enc[l]=getEnc(); for (int l=0;l<NLAY;++l) dec[l]=getDec(); }

  const float *src, *tgt; const int *smask, *tmask;
  if (topSorted) { src=F(idx); smask=(const int*)d_in[idx+1]; tgt=F(idx+2); tmask=(const int*)d_in[idx+3]; }
  else           { src=F(0); tgt=F(1); smask=(const int*)d_in[2]; tmask=(const int*)d_in[3]; }

  // --- workspace layout ---
  size_t cur = 0;
  auto alloc = [&](size_t bytes) -> void* {
    void* r = (char*)d_ws + cur; cur += (bytes + 255) & ~(size_t)255; return r;
  };
  bf16* pwq = (bf16*)alloc((size_t)DM*DM*2);
  bf16* pwk = (bf16*)alloc((size_t)DM*DM*2);
  bf16* pwv = (bf16*)alloc((size_t)DM*DM*2);
  bf16* pwo = (bf16*)alloc((size_t)DM*DM*2);
  bf16* pw1 = (bf16*)alloc((size_t)DM*DFFN*2);
  bf16* pw2 = (bf16*)alloc((size_t)DFFN*DM*2);
  float* encf = (float*)alloc((size_t)ACT_N*4);
  bf16*  encb = (bf16*) alloc((size_t)ACT_N*2);
  float* yf   = (float*)alloc((size_t)ACT_N*4);
  bf16*  yb   = (bf16*) alloc((size_t)ACT_N*2);
  bf16*  qb   = (bf16*) alloc((size_t)ACT_N*2);
  bf16*  kb   = (bf16*) alloc((size_t)ACT_N*2);
  bf16*  vt   = (bf16*) alloc((size_t)ACT_N*2);
  bf16*  ctxb = (bf16*) alloc((size_t)ACT_N*2);
  float* scoresF = (float*)alloc((size_t)NBATCH*NH*TT*TT*4);
  bf16*  probsB  = (bf16*) alloc((size_t)NBATCH*NH*TT*TT*2);
  float* tmpf = (float*)alloc((size_t)ACT_N*4);
  bf16*  h1b  = (bf16*) alloc((size_t)NBATCH*TT*DFFN*2);
  (void)ws_size; (void)out_size; (void)n_in;

  auto gemm = [&](const bf16* A, int lda, const bf16* Bp, int ldb, const float* bias,
                  float* Cf, bf16* Cb, int ldc, int M, int N, int K,
                  int flags, int bm, int om, int batches) {
    dim3 g(N / 64, M / 64, batches);
    gemm_wmma<<<g, dim3(128), 0, stream>>>(A, lda, Bp, ldb, bias, Cf, Cb, ldc,
                                           K, flags, bm, om);
  };
  auto packHS = [&](const float* w, bf16* dst) {
    pack_headstacked<<<(DM * DM) / 256, 256, 0, stream>>>(w, dst);
  };
  auto packT = [&](const float* w, bf16* dst, int K, int N) {
    pack_transpose<<<(K * N) / 256, 256, 0, stream>>>(w, dst, K, N);
  };

  auto runMHA = [&](const bf16* qsrc, const bf16* kvsrc, const int* mask,
                    const MHAp& m, float* outf) {
    packHS(m.wq, pwq); packHS(m.wk, pwk); packHS(m.wv, pwv);
    packT(m.wo, pwo, DM, DM);
    gemm(qsrc,  DM, pwq, DM, m.bq, nullptr, qb, DM, NBATCH*TT, DM, DM, FLAG_BF, BM_NONE, OM_LIN, 1);
    gemm(kvsrc, DM, pwk, DM, m.bk, nullptr, kb, DM, NBATCH*TT, DM, DM, FLAG_BF, BM_NONE, OM_LIN, 1);
    gemm(kvsrc, DM, pwv, DM, m.bv, nullptr, vt, 0,  NBATCH*TT, DM, DM, FLAG_BF, BM_NONE, OM_VT, 1);
    // scores[b,h,t,s] = q . k  (per-head batched)
    gemm(qb, DM, kb, DM, nullptr, scoresF, nullptr, TT, TT, TT, HDM, FLAG_F32, BM_QK, OM_LIN, NBATCH*NH);
    softmax_bf16<<<NBATCH*NH*TT, 256, 0, stream>>>(scoresF, probsB, mask);
    // context = probs x V, scattered back to [B,T,H*HD]
    gemm(probsB, TT, vt, TT, nullptr, nullptr, ctxb, 0, TT, HDM, TT, FLAG_BF, BM_PV, OM_CTX, NBATCH*NH);
    gemm(ctxb, DM, pwo, DM, m.bo, outf, nullptr, DM, NBATCH*TT, DM, DM, FLAG_F32, BM_NONE, OM_LIN, 1);
  };
  auto runFFN = [&](const bf16* xin, const FFNp& f, float* outf) {
    packT(f.w1, pw1, DM, DFFN); packT(f.w2, pw2, DFFN, DM);
    gemm(xin, DM, pw1, DM, f.b1, nullptr, h1b, DFFN, NBATCH*TT, DFFN, DM,
         FLAG_BF | FLAG_RELU, BM_NONE, OM_LIN, 1);
    gemm(h1b, DFFN, pw2, DFFN, f.b2, outf, nullptr, DM, NBATCH*TT, DM, DFFN,
         FLAG_F32, BM_NONE, OM_LIN, 1);
  };
  auto runLN = [&](const float* resid, const float* upd, const LNp& l,
                   float* of, bf16* ob, int mul) {
    ln_fused<<<NBATCH * TT, 256, 0, stream>>>(resid, upd, l.g, l.b, of, ob, mul);
  };

  // ---------------- Encoder ----------------
  cvt_copy<<<ACT_N / 256, 256, 0, stream>>>(src, encf, encb, ACT_N);
  for (int l = 0; l < NLAY; ++l) {
    runMHA(encb, encb, smask, enc[l].mha, tmpf);
    runLN(encf, tmpf, enc[l].ln1, encf, encb, /*mul=*/1);   // x = LN(x * mha(x))
    runFFN(encb, enc[l].ffn, tmpf);
    runLN(encf, tmpf, enc[l].ln2, encf, encb, 0);
  }

  // ---------------- Decoder ----------------
  cvt_copy<<<ACT_N / 256, 256, 0, stream>>>(tgt, yf, yb, ACT_N);
  for (int l = 0; l < NLAY; ++l) {
    runMHA(yb, yb, tmask, dec[l].mha1, tmpf);
    runLN(yf, tmpf, dec[l].ln1, yf, yb, 0);
    runMHA(yb, encb, smask, dec[l].mha2, tmpf);             // k,v from encoder out
    runLN(yf, tmpf, dec[l].ln2, yf, yb, 0);
    runFFN(yb, dec[l].ffn, tmpf);
    float* outDst = (l == NLAY - 1) ? (float*)d_out : yf;
    runLN(yf, tmpf, dec[l].ln3, outDst, yb, 0);
  }
}
